// LSTM_31293131719234
// MI455X (gfx1250) — matrix operand order repaired and checked by reading, not compiled, not run on hardware
//
#include <hip/hip_runtime.h>
#include <hip/hip_bf16.h>

// ---------------------------------------------------------------------------
// LSTM forward on gfx1250: fused [x_t | h_{t-1}] @ [Wx ; Wh] recurrence using
// v_wmma_f32_16x16x32_bf16, fp32 cell state, bf16 hidden-state ping-pong.
// Each wave: 32 rows x 16 cols x 4 gates -> 8 WMMAs per 32-K chunk.
// ---------------------------------------------------------------------------

typedef __bf16 bf16;
typedef __attribute__((ext_vector_type(16))) __bf16 v16bf;
typedef __attribute__((ext_vector_type(8)))  __bf16 bf16x8;
typedef __attribute__((ext_vector_type(8)))  float  v8f;

constexpr int Bsz  = 512;   // batch
constexpr int Seq  = 512;   // sequence length
constexpr int Fdim = 512;   // input size
constexpr int Hdim = 512;   // hidden size
constexpr int KTOT = Fdim + Hdim;   // 1024 fused K
constexpr int GW   = 4 * Hdim;      // 2048 gate width

__device__ __forceinline__ float sigmoid_fast(float x) {
  return 1.0f / (1.0f + __expf(-x));
}
__device__ __forceinline__ float tanh_fast(float x) {
  return 1.0f - 2.0f / (__expf(2.0f * x) + 1.0f);
}
__device__ __forceinline__ v16bf cat8(bf16x8 lo, bf16x8 hi) {
  return __builtin_shufflevector(lo, hi,
      0, 1, 2, 3, 4, 5, 6, 7, 8, 9, 10, 11, 12, 13, 14, 15);
}

// --------------------------------------------------------------------------
// Init: WcatT[col][k] = bf16( k<F ? Wx[k][col] : Wh[k-F][col] ), col-major so
// WMMA B-fragments are contiguous-K loads. Also zero h0 (bf16) and c (f32).
// --------------------------------------------------------------------------
__global__ void lstm_init(const float* __restrict__ Wx, const float* __restrict__ Wh,
                          bf16* __restrict__ WcatT, bf16* __restrict__ h0,
                          float* __restrict__ c0) {
  int idx = blockIdx.x * blockDim.x + threadIdx.x;
  if (idx < GW * KTOT) {
    int col = idx >> 10;          // 0..2047
    int k   = idx & (KTOT - 1);   // 0..1023
    float w = (k < Fdim) ? Wx[(size_t)k * GW + col]
                         : Wh[(size_t)(k - Fdim) * GW + col];
    WcatT[idx] = (bf16)w;
  }
  if (idx < Bsz * Hdim) {
    h0[idx] = (bf16)0.0f;
    c0[idx] = 0.0f;
  }
}

// --------------------------------------------------------------------------
// One 32-K MMA chunk: 2 A fragments (two 16-row subtiles) x 4 gate B
// fragments -> 8 WMMAs. A from LDS, B straight from L2-resident WcatT.
// --------------------------------------------------------------------------
__device__ __forceinline__ void mma_chunk(const bf16* __restrict__ As,
                                          const bf16* __restrict__ WcatT,
                                          int kglob, int waveM, int ncol0,
                                          int lhalf, int ln16, v8f acc[4][2]) {
  // A fragment (16x32 bf16): lane holds row, K = lhalf*16 + e (contiguous)
  v16bf af[2];
#pragma unroll
  for (int m = 0; m < 2; ++m) {
    const bf16* ap = &As[(waveM + m * 16 + ln16) * 32 + lhalf * 16];
    af[m] = cat8(*(const bf16x8*)(ap), *(const bf16x8*)(ap + 8));
  }
  // B fragment (32x16 bf16): lane holds col, K = lhalf*16 + e (contiguous)
  v16bf bfv[4];
#pragma unroll
  for (int g = 0; g < 4; ++g) {
    const bf16* bp = WcatT + (size_t)(g * Hdim + ncol0) * KTOT + kglob + lhalf * 16;
    bfv[g] = cat8(*(const bf16x8*)(bp), *(const bf16x8*)(bp + 8));
  }
#pragma unroll
  for (int g = 0; g < 4; ++g)
#pragma unroll
    for (int m = 0; m < 2; ++m)
      acc[g][m] = __builtin_amdgcn_wmma_f32_16x16x32_bf16(
          false, af[m], false, bfv[g], (short)0, acc[g][m], false, false);
}

// --------------------------------------------------------------------------
// One LSTM time step. Grid = (B/64, H/64), 256 threads (8 waves).
// Wave wv: rows (wv>>2)*32 .. +32, cols (wv&3)*16 .. +16 (per gate).
// --------------------------------------------------------------------------
__global__ __launch_bounds__(256)
void lstm_step(const float* __restrict__ x, const bf16* __restrict__ WcatT,
               const float* __restrict__ bx, const float* __restrict__ bh,
               const bf16* __restrict__ h_in, bf16* __restrict__ h_out,
               float* __restrict__ c, float* __restrict__ hc_out, int t) {
  __shared__ __align__(16) bf16 As[64 * 32];   // 64 rows x 32 K, 4 KB

  const int tid   = threadIdx.x;
  const int lane  = tid & 31;
  const int wv    = tid >> 5;
  const int lhalf = lane >> 4;     // 0: lanes 0-15, 1: lanes 16-31
  const int ln16  = lane & 15;
  const int blockM = blockIdx.x * 64;
  const int blockN = blockIdx.y * 64;
  const int waveM  = (wv >> 2) * 32;              // 0 or 32 (As-local rows)
  const int ncol0  = blockN + (wv & 3) * 16;      // base hidden column

  v8f acc[4][2];
#pragma unroll
  for (int g = 0; g < 4; ++g)
#pragma unroll
    for (int m = 0; m < 2; ++m)
      acc[g][m] = (v8f){0.f, 0.f, 0.f, 0.f, 0.f, 0.f, 0.f, 0.f};

  // staging: each thread loads 8 elements of the 64x32 A tile
  const int srow = tid >> 2;          // 0..63
  const int skb  = (tid & 3) * 8;     // 0,8,16,24
  const int gm   = blockM + srow;     // global batch row
  const float* xrow = x + ((size_t)gm * Seq + t) * Fdim;
  const bf16*  hrow = h_in + (size_t)gm * Hdim;

  // -------- phase 1: K in [0, F) from x_t (fp32 -> bf16 via LDS) ----------
  for (int kk = 0; kk < Fdim; kk += 32) {
    __syncthreads();
    float4 f0 = *(const float4*)(xrow + kk + skb);
    float4 f1 = *(const float4*)(xrow + kk + skb + 4);
    if (kk + 32 < Fdim)
      __builtin_prefetch(xrow + kk + 32 + skb, 0, 1);   // global_prefetch_b8
    bf16x8 av;
    av[0] = (bf16)f0.x; av[1] = (bf16)f0.y; av[2] = (bf16)f0.z; av[3] = (bf16)f0.w;
    av[4] = (bf16)f1.x; av[5] = (bf16)f1.y; av[6] = (bf16)f1.z; av[7] = (bf16)f1.w;
    *(bf16x8*)(&As[srow * 32 + skb]) = av;
    __syncthreads();
    mma_chunk(As, WcatT, kk, waveM, ncol0, lhalf, ln16, acc);
  }

  // -------- phase 2: K in [F, F+H) from h_{t-1} (already bf16) ------------
  for (int kk = 0; kk < Hdim; kk += 32) {
    __syncthreads();
    *(bf16x8*)(&As[srow * 32 + skb]) = *(const bf16x8*)(hrow + kk + skb);
    __syncthreads();
    mma_chunk(As, WcatT, Fdim + kk, waveM, ncol0, lhalf, ln16, acc);
  }

  // ---- fused gate nonlinearity + state update (in-register) --------------
  const int col = ncol0 + ln16;
  const float bi = bx[0 * Hdim + col] + bh[0 * Hdim + col];
  const float bf = bx[1 * Hdim + col] + bh[1 * Hdim + col];
  const float bg = bx[2 * Hdim + col] + bh[2 * Hdim + col];
  const float bo = bx[3 * Hdim + col] + bh[3 * Hdim + col];

#pragma unroll
  for (int m = 0; m < 2; ++m) {
#pragma unroll
    for (int j = 0; j < 8; ++j) {
      // C/D layout: VGPR j -> row = lhalf*8 + j, col = lane&15
      const int row = blockM + waveM + m * 16 + lhalf * 8 + j;
      const size_t idx = (size_t)row * Hdim + col;
      const float it = sigmoid_fast(acc[0][m][j] + bi);
      const float ft = sigmoid_fast(acc[1][m][j] + bf);
      const float gt = tanh_fast   (acc[2][m][j] + bg);
      const float ot = sigmoid_fast(acc[3][m][j] + bo);
      const float cn = ft * c[idx] + it * gt;
      c[idx] = cn;
      const float hn = ot * tanh_fast(cn);
      h_out[idx] = (bf16)hn;
      if (hc_out) {               // last step only: emit (h, c) fp32
        hc_out[idx] = hn;
        hc_out[(size_t)Bsz * Hdim + idx] = cn;
      }
    }
  }
}

// --------------------------------------------------------------------------
extern "C" void kernel_launch(void* const* d_in, const int* in_sizes, int n_in,
                              void* d_out, int out_size, void* d_ws, size_t ws_size,
                              hipStream_t stream) {
  (void)in_sizes; (void)n_in; (void)out_size; (void)ws_size;
  const float* x  = (const float*)d_in[0];
  const float* Wx = (const float*)d_in[1];
  const float* Wh = (const float*)d_in[2];
  const float* bx = (const float*)d_in[3];
  const float* bh = (const float*)d_in[4];

  // workspace layout (~6 MB total)
  char* ws = (char*)d_ws;
  bf16*  WcatT = (bf16*)ws;                                   // 2048*1024*2 = 4 MB
  bf16*  hbuf0 = (bf16*)(ws + (4u << 20));                    // 512 KB
  bf16*  hbuf1 = (bf16*)(ws + (4u << 20) + (512u << 10));     // 512 KB
  float* cbuf  = (float*)(ws + (5u << 20));                   // 1 MB

  lstm_init<<<(GW * KTOT + 255) / 256, 256, 0, stream>>>(Wx, Wh, WcatT, hbuf0, cbuf);

  dim3 grid(Bsz / 64, Hdim / 64);
  for (int t = 0; t < Seq; ++t) {
    bf16* hin  = (t & 1) ? hbuf1 : hbuf0;
    bf16* hout = (t & 1) ? hbuf0 : hbuf1;
    float* hc  = (t == Seq - 1) ? (float*)d_out : nullptr;
    lstm_step<<<grid, 256, 0, stream>>>(x, WcatT, bx, bh, hin, hout, cbuf, hc, t);
  }
}
